// FCRelationalEmbedding_58282706207120
// MI455X (gfx1250) — matrix operand order repaired and checked by reading, not compiled, not run on hardware
//
#include <hip/hip_runtime.h>

// Problem constants from the reference
#define NN   768
#define UU   64
#define FEAT 136
#define KPAD 160   // FEAT padded to 5 K-tiles of 32 (bf16 WMMA)
#define NPAD 144   // FEAT padded to 9 N-tiles of 16
#define BB   2
#define ICH  16    // i-dimension chunks (deterministic reduction)
#define RPC  (NN / ICH)  // 48 rows per chunk
#define SLOPE 0.01f

typedef __attribute__((ext_vector_type(16))) __bf16 v16bf;
typedef __attribute__((ext_vector_type(8)))  float  v8f;

struct __align__(16) U4 { unsigned int x, y, z, w; };
union Frag { v16bf v; U4 q[2]; };

// ---------------- prep kernels ----------------

__global__ void zero_f32(float* p, int n) {
  int i = blockIdx.x * blockDim.x + threadIdx.x;
  if (i < n) p[i] = 0.f;
}

// mask = (sum_k rel > 0); relbf = bf16(rel); D[i] += mask (exact integer adds)
__global__ void maskrel_kernel(const float* __restrict__ rel,
                               float* __restrict__ maskF,
                               __bf16* __restrict__ relbf,
                               float* __restrict__ Dacc) {
  int idx = blockIdx.x * blockDim.x + threadIdx.x;  // over NN*NN pairs
  if (idx >= NN * NN) return;
  const float* rp = rel + (size_t)idx * 8;
  float sum = 0.f;
  __bf16 tmp[8];
#pragma unroll
  for (int k = 0; k < 8; ++k) { float v = rp[k]; sum += v; tmp[k] = (__bf16)v; }
  float m = (sum > 0.f) ? 1.f : 0.f;
  maskF[idx] = m;
#pragma unroll
  for (int k = 0; k < 8; ++k) relbf[(size_t)idx * 8 + k] = tmp[k];
  if (m > 0.f) atomicAdd(&Dacc[idx / NN], 1.f);  // 0/1 adds: order-independent
}

__global__ void dinv_kernel(const float* __restrict__ D, float* __restrict__ Dinv, int n) {
  int i = blockIdx.x * blockDim.x + threadIdx.x;
  if (i < n) Dinv[i] = 1.f / D[i];
}

// Pack W1 (stored [k][n], k-major) into transposed, padded bf16 W1T[n][k];
// pad b1 / W2 to NPAD floats.
__global__ void packw_kernel(const float* __restrict__ W1, const float* __restrict__ b1,
                             const float* __restrict__ W2, const float* __restrict__ b2,
                             __bf16* __restrict__ W1T, float* __restrict__ b1p,
                             float* __restrict__ W2p, float* __restrict__ b2p) {
  int idx = blockIdx.x * blockDim.x + threadIdx.x;  // over NPAD*KPAD
  if (idx < NPAD * KPAD) {
    int n = idx / KPAD, k = idx % KPAD;
    float v = (n < FEAT && k < FEAT) ? W1[k * FEAT + n] : 0.f;
    W1T[idx] = (__bf16)v;
  }
  if (idx < NPAD) {
    b1p[idx] = (idx < FEAT) ? b1[idx] : 0.f;
    W2p[idx] = (idx < FEAT) ? W2[idx] : 0.f;
  }
  if (idx == 0) b2p[0] = b2[0];
}

__global__ void tobf_kernel(const float* __restrict__ x, __bf16* __restrict__ xb, int n) {
  int i = blockIdx.x * blockDim.x + threadIdx.x;
  if (i < n) xb[i] = (__bf16)x[i];
}

// ---------------- hop kernel: per-edge MLP via WMMA ----------------
// One wave (32 threads) per block. Block = (j-tile of 16, batch, i-chunk).
// For each row i in its chunk: M=16 pairs (i, j0..j0+15), K=160 features,
// N=144 layer-1 outputs -> 9 x 5 = 45 v_wmma_f32_16x16x32_bf16 per active row.
__global__ __launch_bounds__(32)
void hop_kernel(const __bf16* __restrict__ xbf, const __bf16* __restrict__ relbf,
                const float* __restrict__ maskF, const float* __restrict__ Dinv,
                const __bf16* __restrict__ W1T, const float* __restrict__ b1p,
                const float* __restrict__ W2p, const float* __restrict__ b2p,
                float* __restrict__ partial) {
  __shared__ __align__(16) __bf16 sW[NPAD * KPAD];  // 45 KB transposed weights

  const int lane = threadIdx.x;
  // stage W1T -> LDS (2880 x 16B, uniform trip count: no divergence)
  {
    const U4* src = (const U4*)W1T;
    U4* dst = (U4*)sW;
#pragma unroll 1
    for (int idx = lane; idx < (NPAD * KPAD * 2) / 16; idx += 32) dst[idx] = src[idx];
  }
  __syncthreads();

  const int nlo = lane & 15;          // N-column / j-offset within tile
  const int hb  = (lane >> 4) & 1;    // half-wave selector (A: K-half, C: M-half)
  const int b   = blockIdx.y;
  const int j0  = blockIdx.x * 16;
  const int i0  = blockIdx.z * RPC;

  // per-lane bias / layer-2 weight for each N-tile (N = t*16 + nlo)
  float biasv[9], w2v[9];
#pragma unroll
  for (int t = 0; t < 9; ++t) { int nn = t * 16 + nlo; biasv[t] = b1p[nn]; w2v[t] = W2p[nn]; }
  const float b2v = b2p[0];

  const U4* xj = (const U4*)(xbf + (size_t)(b * NN + j0 + nlo) * UU);  // this lane's x_j row

  float acc[8];
#pragma unroll
  for (int r = 0; r < 8; ++r) acc[r] = 0.f;

  for (int i = i0; i < i0 + RPC; ++i) {
    float mval = maskF[(size_t)i * NN + j0 + nlo];
    unsigned long long bal = __ballot(mval > 0.f);
    if ((bal & 0xFFFFull) == 0ull) continue;  // wave-uniform skip of empty tiles
    float di = Dinv[i];

    // A fragments: 16x32 bf16 per ISA layout. Lane<16: K{0..7,16..23}; lane>=16: +8.
    // K 0..63 = x_i (broadcast), 64..127 = x_j(lane), 128..135 = rel_ij, rest zero.
    Frag A[5];
    const U4* xi = (const U4*)(xbf + (size_t)(b * NN + i) * UU);
    A[0].q[0] = xi[hb];     A[0].q[1] = xi[hb + 2];
    A[1].q[0] = xi[hb + 4]; A[1].q[1] = xi[hb + 6];
    A[2].q[0] = xj[hb];     A[2].q[1] = xj[hb + 2];
    A[3].q[0] = xj[hb + 4]; A[3].q[1] = xj[hb + 6];
    U4 z; z.x = z.y = z.z = z.w = 0u;
    U4 r0 = ((const U4*)(relbf + ((size_t)i * NN + j0 + nlo) * 8))[0];
    A[4].q[0] = hb ? z : r0;   // K 128..135 real, 136..159 zero-pad
    A[4].q[1] = z;

    float p2[8];
#pragma unroll
    for (int r = 0; r < 8; ++r) p2[r] = 0.f;

#pragma unroll
    for (int t = 0; t < 9; ++t) {
      v8f c = {0.f, 0.f, 0.f, 0.f, 0.f, 0.f, 0.f, 0.f};
#pragma unroll
      for (int kk = 0; kk < 5; ++kk) {
        // B fragment: lane (hb,nlo) holds W1T[n0+nlo][kk*32 + hb*8 .. +7, kk*32 + 16 + hb*8 .. +7] (packed pairs)
        Frag bw;
        const U4* wp = (const U4*)(sW + (size_t)(t * 16 + nlo) * KPAD + kk * 32 + hb * 8);
        bw.q[0] = wp[0]; bw.q[1] = wp[2];
        c = __builtin_amdgcn_wmma_f32_16x16x32_bf16(false, A[kk].v, false, bw.v,
                                                    (short)0, c, false, false);
      }
      // D layout: vgpr r -> M = hb*8 + r; N = t*16 + nlo. Bias + leaky + layer-2 dot.
#pragma unroll
      for (int r = 0; r < 8; ++r) {
        float h = c[r] + biasv[t];
        h = (h > 0.f) ? h : SLOPE * h;
        p2[r] = fmaf(h, w2v[t], p2[r]);
      }
    }

    // reduce layer-2 partials over the 16 lanes of each half (N dimension)
#pragma unroll
    for (int r = 0; r < 8; ++r) {
      float v = p2[r];
      v += __shfl_xor(v, 1);
      v += __shfl_xor(v, 2);
      v += __shfl_xor(v, 4);
      v += __shfl_xor(v, 8);
      float s = v + b2v;
      s = (s > 0.f) ? s : SLOPE * s;           // leaky
      int srcl = r + ((lane & 16) >> 1);       // lane holding mask for M = hb*8+r
      float mr = __shfl(mval, srcl);
      acc[r] = fmaf(s * mr, di, acc[r]);       // (+= leaky(s)*mask/D[i])
    }
  }

  // lanes 0 and 16 write this chunk's partial column sums for j0+hb*8 .. +7
  if (nlo == 0) {
    float* outp = partial + ((size_t)blockIdx.z * BB + b) * NN + j0 + hb * 8;
#pragma unroll
    for (int r = 0; r < 8; ++r) outp[r] = acc[r];
  }
}

// colsum = fixed-order sum over chunks; x_next = colsum * x_in (and bf16 copy)
__global__ void apply_kernel(const float* __restrict__ partial, const float* __restrict__ xin,
                             float* __restrict__ xout, __bf16* __restrict__ xob) {
  int idx = blockIdx.x * blockDim.x + threadIdx.x;  // over BB*NN
  if (idx >= BB * NN) return;
  float cs = 0.f;
#pragma unroll
  for (int c = 0; c < ICH; ++c) cs += partial[(size_t)c * BB * NN + idx];
  const float* xi = xin + (size_t)idx * UU;
  float* xo = xout + (size_t)idx * UU;
  for (int u = 0; u < UU; ++u) {
    float v = cs * xi[u];
    xo[u] = v;
    if (xob) xob[(size_t)idx * UU + u] = (__bf16)v;
  }
}

// ---------------- launcher ----------------

extern "C" void kernel_launch(void* const* d_in, const int* in_sizes, int n_in,
                              void* d_out, int out_size, void* d_ws, size_t ws_size,
                              hipStream_t stream) {
  (void)in_sizes; (void)n_in; (void)out_size; (void)ws_size;
  const float* seq = (const float*)d_in[0];
  const float* rel = (const float*)d_in[1];
  const float* w11 = (const float*)d_in[2];
  const float* b11 = (const float*)d_in[3];
  const float* w12 = (const float*)d_in[4];
  const float* b12 = (const float*)d_in[5];
  const float* w21 = (const float*)d_in[6];
  const float* b21 = (const float*)d_in[7];
  const float* w22 = (const float*)d_in[8];
  const float* b22 = (const float*)d_in[9];

  char* ws = (char*)d_ws;
  size_t off = 0;
  auto carve = [&](size_t bytes) -> void* {
    void* p = ws + off;
    off += (bytes + 255) & ~(size_t)255;
    return p;
  };
  float*  maskF  = (float*)carve((size_t)NN * NN * 4);
  __bf16* relbf  = (__bf16*)carve((size_t)NN * NN * 8 * 2);
  float*  Dbuf   = (float*)carve(NN * 4);
  float*  Dinv   = (float*)carve(NN * 4);
  __bf16* W1T1   = (__bf16*)carve(NPAD * KPAD * 2);
  __bf16* W1T2   = (__bf16*)carve(NPAD * KPAD * 2);
  float*  b1p1   = (float*)carve(NPAD * 4);
  float*  W2p1   = (float*)carve(NPAD * 4);
  float*  b2p1   = (float*)carve(4);
  float*  b1p2   = (float*)carve(NPAD * 4);
  float*  W2p2   = (float*)carve(NPAD * 4);
  float*  b2p2   = (float*)carve(4);
  __bf16* xbf1   = (__bf16*)carve((size_t)BB * NN * UU * 2);
  float*  x1     = (float*)carve((size_t)BB * NN * UU * 4);
  __bf16* xbf2   = (__bf16*)carve((size_t)BB * NN * UU * 2);
  float*  partial = (float*)carve((size_t)ICH * BB * NN * 4);

  zero_f32<<<(NN + 255) / 256, 256, 0, stream>>>(Dbuf, NN);
  maskrel_kernel<<<(NN * NN) / 256, 256, 0, stream>>>(rel, maskF, relbf, Dbuf);
  dinv_kernel<<<(NN + 255) / 256, 256, 0, stream>>>(Dbuf, Dinv, NN);
  packw_kernel<<<(NPAD * KPAD) / 256, 256, 0, stream>>>(w11, b11, w12, b12, W1T1, b1p1, W2p1, b2p1);
  packw_kernel<<<(NPAD * KPAD) / 256, 256, 0, stream>>>(w21, b21, w22, b22, W1T2, b1p2, W2p2, b2p2);
  tobf_kernel<<<(BB * NN * UU) / 256, 256, 0, stream>>>(seq, xbf1, BB * NN * UU);

  dim3 grid(NN / 16, BB, ICH);   // 48 x 2 x 16 single-wave blocks
  hop_kernel<<<grid, 32, 0, stream>>>(xbf1, relbf, maskF, Dinv, W1T1, b1p1, W2p1, b2p1, partial);
  apply_kernel<<<(BB * NN + 255) / 256, 256, 0, stream>>>(partial, seq, x1, xbf2);
  hop_kernel<<<grid, 32, 0, stream>>>(xbf2, relbf, maskF, Dinv, W1T2, b1p2, W2p2, b2p2, partial);
  apply_kernel<<<(BB * NN + 255) / 256, 256, 0, stream>>>(partial, x1, (float*)d_out, (__bf16*)nullptr);
}
